// JointlyTrainModel_21620865368320
// MI455X (gfx1250) — matrix operand order repaired and checked by reading, not compiled, not run on hardware
//
#include <hip/hip_runtime.h>
#include <stdint.h>

// ---------------------------------------------------------------------------
// ChebConv GNN on MI455X (gfx1250): all matmuls via v_wmma_f32_16x16x32_bf16
// (bf16 A/B, fp32 accumulate).  Graph topology is identical across the 2048
// batched graphs, so prop() == dense 62x62 Laplacian matmul.
// GEMM staging uses gfx1250 async global->LDS copies (ASYNCcnt path);
// B-operands always come from K-major buffers so every fragment is built
// from wide aligned loads (no 16-bit packing).
// ---------------------------------------------------------------------------

#define NODES 62
#define NODESP 64                 // padded
#define BATCHG 2048
#define NTOT (NODES * BATCHG)     // 126976
#define INFEAT 128
#define OUTC 62
#define FEAT 248                  // 4*62, per-node concat of x1..x4
#define FLATK 15376               // 62*248, FC1 K
#define FLATKP 15392              // padded to /32
#define LIN1 512
#define LIN2 256
#define BNEPS 1e-5f

typedef __attribute__((ext_vector_type(16))) __bf16 bf16x16;
typedef __attribute__((ext_vector_type(8)))  __bf16 bf16x8;
typedef __attribute__((ext_vector_type(8)))  float  f32x8;

__device__ static inline f32x8 fzero8() {
  f32x8 z;
#pragma unroll
  for (int i = 0; i < 8; ++i) z[i] = 0.0f;
  return z;
}

__device__ static inline f32x8 wmma_bf16(bf16x16 a, bf16x16 b, f32x8 c) {
  // (neg_a, A, neg_b, B, c_mod, C, reuse_a, reuse_b)
  return __builtin_amdgcn_wmma_f32_16x16x32_bf16(false, a, false, b,
                                                 (short)0, c, false, false);
}

// Async 16B global->LDS copy (GLOBAL_LOAD_ASYNC_TO_LDS_B128, ASYNCcnt).
// lds_addr: wave-relative LDS byte address == low 32 bits of generic pointer
// (ISA 10.2: LDS flat address = addr[31:0]).
__device__ static inline void async_copy_b128(unsigned lds_addr,
                                              const void* gptr) {
  asm volatile("global_load_async_to_lds_b128 %0, %1, off"
               :
               : "v"(lds_addr), "v"((unsigned long long)(uintptr_t)gptr)
               : "memory");
}

__device__ static inline void wait_asynccnt0() {
  asm volatile("s_wait_asynccnt 0" ::: "memory");
}

// A fragment (16x32, MxK) from row-major bf16.  Lane half selects
// khalf in {0,16}; k khalf..khalf+15 are contiguous (matches B fragment),
// so the fragment is two aligned 16-byte loads.
__device__ static inline bf16x16 load_a_frag(const __bf16* base, int stride,
                                             int row0, int k0, int lane) {
  int m = lane & 15;
  int khalf = (lane >> 4) * 16;
  const __bf16* p = base + (size_t)(row0 + m) * stride + k0 + khalf;
  bf16x8 lo = *(const bf16x8*)p;
  bf16x8 hi = *(const bf16x8*)(p + 8);
  bf16x16 a;
#pragma unroll
  for (int v = 0; v < 8; ++v) {
    a[v] = lo[v];
    a[v + 8] = hi[v];
  }
  return a;
}

// B fragment (32x16, KxN) from K-major Bt[N][ldk] (B[k][n] = Bt[n][k]):
// one aligned 32-byte load per lane.
__device__ static inline bf16x16 load_bt_frag(const __bf16* bt, int ldk,
                                              int k0, int n0, int lane) {
  int n = n0 + (lane & 15);
  int kb = k0 + (lane >> 4) * 16;
  return *(const bf16x16*)(bt + (size_t)n * ldk + kb);
}

// ---------------------------------------------------------------------------
// prep kernels
// ---------------------------------------------------------------------------
__global__ void zero_f32(float* p, int n) {
  for (int i = blockIdx.x * blockDim.x + threadIdx.x; i < n;
       i += gridDim.x * blockDim.x)
    p[i] = 0.0f;
}

__global__ void deg_accum(const int* __restrict__ ei, float* __restrict__ deg,
                          int E) {
  for (int e = blockIdx.x * blockDim.x + threadIdx.x; e < E;
       e += gridDim.x * blockDim.x)
    atomicAdd(&deg[ei[e]], 1.0f);
}

// Build shared 62x62 scaled Laplacian (padded 64x64) from graph-0 edges.
__global__ void lhat_build(const int* __restrict__ ei,
                           const float* __restrict__ deg,
                           float* __restrict__ lhat, int E) {
  for (int e = blockIdx.x * blockDim.x + threadIdx.x; e < E;
       e += gridDim.x * blockDim.x) {
    int r = ei[e], c = ei[E + e];
    if (r < NODES && c < NODES) {
      float dr = deg[r] > 0.0f ? rsqrtf(deg[r]) : 0.0f;
      float dc = deg[c] > 0.0f ? rsqrtf(deg[c]) : 0.0f;
      atomicAdd(&lhat[r * 64 + c], -(dr * dc));
    }
  }
}

// fp32 W[Bk][K][N] -> bf16 Wt[Bk][Npad][Kpad], zero padded in both dims.
__global__ void wprep(const float* __restrict__ w, __bf16* __restrict__ wt,
                      int Bk, int K, int Nn, int Npad, int Kpad) {
  size_t total = (size_t)Bk * Npad * Kpad;
  for (size_t i = (size_t)blockIdx.x * blockDim.x + threadIdx.x; i < total;
       i += (size_t)gridDim.x * blockDim.x) {
    int kk = (int)(i % Kpad);
    size_t rem = i / Kpad;
    int n = (int)(rem % Npad);
    int bk = (int)(rem / Npad);
    float v = (kk < K && n < Nn) ? w[((size_t)bk * K + kk) * Nn + n] : 0.0f;
    wt[i] = (__bf16)v;
  }
}

// xcat flat [2048][15376] -> featp [2048][15392] with zero K tail (16B vec).
__global__ void pad_feats(const __bf16* __restrict__ src,
                          __bf16* __restrict__ dst) {
  size_t total = (size_t)BATCHG * (FLATKP / 8);
  for (size_t i = (size_t)blockIdx.x * blockDim.x + threadIdx.x; i < total;
       i += (size_t)gridDim.x * blockDim.x) {
    int kc = (int)(i % (FLATKP / 8));
    size_t row = i / (FLATKP / 8);
    bf16x8 v;
    if (kc * 8 < FLATK) {
      v = *(const bf16x8*)(src + row * FLATK + kc * 8);
    } else {
#pragma unroll
      for (int j = 0; j < 8; ++j) v[j] = (__bf16)0.0f;
    }
    *(bf16x8*)(dst + row * FLATKP + kc * 8) = v;
  }
}

// ---------------------------------------------------------------------------
// Fused Cheb layer: one workgroup per graph (62 nodes, padded to 64).
// out = relu( H W0 + T1 W1 + T2 W2 + b ),  T1 = L H,  T2 = 2 L T1 - H
// Processed in 64-feature chunks (the Chebyshev recursion is column-wise
// independent).  sH/sT hold row-major [node][feat] (A-operand orientation);
// sHt/sTt hold col-major [feat][node] (B-operand orientation).
// Zero-padded L columns / W rows annihilate pad garbage exactly.
// ---------------------------------------------------------------------------
__global__ __launch_bounds__(256) void cheb_kernel(
    const float* __restrict__ x,        // [NTOT,128] fp32
    __bf16* xcat,                       // [NTOT,248] bf16
    const float* __restrict__ lhat,     // [64*64] fp32 padded
    const __bf16* __restrict__ wt,      // [3][64][kpad] bf16, K-major
    const float* __restrict__ bias,     // [62]
    int kpad, int colbase) {
  __shared__ __align__(64) __bf16 sL[64 * 64];
  __shared__ __align__(64) __bf16 sH[64 * 64];   // H, later T2 (row-major)
  __shared__ __align__(64) __bf16 sHt[64 * 64];  // H transposed
  __shared__ __align__(64) __bf16 sT[64 * 64];   // T1 (row-major)
  __shared__ __align__(64) __bf16 sTt[64 * 64];  // T1 transposed
  int tid = threadIdx.x, lane = tid & 31, w = tid >> 5;
  int nodebase = blockIdx.x * NODES;

  for (int i = tid; i < 64 * 64; i += 256) sL[i] = (__bf16)lhat[i];

  f32x8 acc[2] = {fzero8(), fzero8()};
  int nchunks = kpad / 64;

  for (int c = 0; c < nchunks; ++c) {
    __syncthreads();  // protect sH/sHt/sT/sTt rewrite vs previous-chunk reads
    if (c < 2) {
      // 64 fp32 features per row from x: float4 loads, cvt, write both copies
      for (int i = tid; i < 64 * 16; i += 256) {
        int r = i >> 4, part = i & 15;
        int node = nodebase + (r < NODES ? r : NODES - 1);  // clamp pad rows
        float4 f = *(const float4*)(x + (size_t)node * INFEAT + c * 64 + part * 4);
        __bf16 e0 = (__bf16)f.x, e1 = (__bf16)f.y, e2 = (__bf16)f.z,
               e3 = (__bf16)f.w;
        __bf16* d = sH + r * 64 + part * 4;
        d[0] = e0;
        d[1] = e1;
        d[2] = e2;
        d[3] = e3;
        __bf16* dt = sHt + part * 4 * 64 + r;
        dt[0] = e0;
        dt[64] = e1;
        dt[128] = e2;
        dt[192] = e3;
      }
    } else {
      // 64 bf16 features per row from xcat: 16B vector + transposed scatter
      int cb = c * 64 - INFEAT;
      for (int i = tid; i < 64 * 8; i += 256) {
        int r = i >> 3, part = i & 7;
        int node = nodebase + (r < NODES ? r : NODES - 1);
        bf16x8 v = *(const bf16x8*)(xcat + (size_t)node * FEAT + cb + part * 8);
        *(bf16x8*)(sH + r * 64 + part * 8) = v;
        __bf16* dt = sHt + part * 8 * 64 + r;
#pragma unroll
        for (int j = 0; j < 8; ++j) dt[j * 64] = v[j];
      }
    }
    __syncthreads();

    // phase 2: T1 = L*Hc -> sT/sTt ; acc += Hc * W0[chunk]
#pragma unroll
    for (int tt = 0; tt < 2; ++tt) {
      int t = 2 * w + tt, mt = t >> 2, nt = t & 3;
      f32x8 tacc = fzero8();
#pragma unroll
      for (int ks = 0; ks < 2; ++ks) {
        bf16x16 a = load_a_frag(sL, 64, mt * 16, ks * 32, lane);
        bf16x16 b = load_bt_frag(sHt, 64, ks * 32, nt * 16, lane);
        tacc = wmma_bf16(a, b, tacc);
      }
      int col = nt * 16 + (lane & 15);
#pragma unroll
      for (int r8 = 0; r8 < 8; ++r8) {
        int row = mt * 16 + r8 + ((lane >> 4) << 3);
        __bf16 tv = (__bf16)tacc[r8];
        sT[row * 64 + col] = tv;
        sTt[col * 64 + row] = tv;
      }
#pragma unroll
      for (int ks = 0; ks < 2; ++ks) {
        bf16x16 a = load_a_frag(sH, 64, mt * 16, ks * 32, lane);
        bf16x16 b = load_bt_frag(wt, kpad, c * 64 + ks * 32, nt * 16, lane);
        acc[tt] = wmma_bf16(a, b, acc[tt]);
      }
    }
    __syncthreads();

    // phase 3: acc += T1 * W1[chunk] ; T2 = 2*L*T1 - Hc -> overwrite sH
#pragma unroll
    for (int tt = 0; tt < 2; ++tt) {
      int t = 2 * w + tt, mt = t >> 2, nt = t & 3;
#pragma unroll
      for (int ks = 0; ks < 2; ++ks) {
        bf16x16 a = load_a_frag(sT, 64, mt * 16, ks * 32, lane);
        bf16x16 b = load_bt_frag(wt + (size_t)1 * NODESP * kpad, kpad,
                                 c * 64 + ks * 32, nt * 16, lane);
        acc[tt] = wmma_bf16(a, b, acc[tt]);
      }
      f32x8 tacc = fzero8();
#pragma unroll
      for (int ks = 0; ks < 2; ++ks) {
        bf16x16 a = load_a_frag(sL, 64, mt * 16, ks * 32, lane);
        bf16x16 b = load_bt_frag(sTt, 64, ks * 32, nt * 16, lane);
        tacc = wmma_bf16(a, b, tacc);
      }
      int col = nt * 16 + (lane & 15);
#pragma unroll
      for (int r8 = 0; r8 < 8; ++r8) {
        int row = mt * 16 + r8 + ((lane >> 4) << 3);
        float hv = (float)sH[row * 64 + col];
        sH[row * 64 + col] = (__bf16)(2.0f * tacc[r8] - hv);
      }
    }
    __syncthreads();

    // phase 4: acc += T2 * W2[chunk]   (T2 now in sH)
#pragma unroll
    for (int tt = 0; tt < 2; ++tt) {
      int t = 2 * w + tt, mt = t >> 2, nt = t & 3;
#pragma unroll
      for (int ks = 0; ks < 2; ++ks) {
        bf16x16 a = load_a_frag(sH, 64, mt * 16, ks * 32, lane);
        bf16x16 b = load_bt_frag(wt + (size_t)2 * NODESP * kpad, kpad,
                                 c * 64 + ks * 32, nt * 16, lane);
        acc[tt] = wmma_bf16(a, b, acc[tt]);
      }
    }
  }

  // epilogue: bias + relu -> xcat columns [colbase, colbase+62)
#pragma unroll
  for (int tt = 0; tt < 2; ++tt) {
    int t = 2 * w + tt, mt = t >> 2, nt = t & 3;
    int col = nt * 16 + (lane & 15);
    if (col < OUTC) {
      float bv = bias[col];
#pragma unroll
      for (int r8 = 0; r8 < 8; ++r8) {
        int row = mt * 16 + r8 + ((lane >> 4) << 3);
        if (row < NODES) {
          float v = fmaxf(acc[tt][r8] + bv, 0.0f);
          xcat[(size_t)(nodebase + row) * FEAT + colbase + col] = (__bf16)v;
        }
      }
    }
  }
}

// ---------------------------------------------------------------------------
// Generic bf16 WMMA GEMM: C[M,N] = A[M,Kp] * Bt[N,Kp]^T, fp32 out.
// Kp multiple of 32, buffers pre-padded -> zero guards.  64x64 macro-tile,
// 8 waves x 2 C-tiles.  Staging via async global->LDS b128 copies
// (ASYNCcnt), one per thread per tile, plus global_prefetch ahead.
// ---------------------------------------------------------------------------
__global__ __launch_bounds__(256) void gemm_bf16(
    const __bf16* __restrict__ A, int lda, const __bf16* __restrict__ Bt,
    int ldk, float* __restrict__ C, int M, int N, int Kp) {
  __shared__ __align__(64) __bf16 sA[64 * 32];
  __shared__ __align__(64) __bf16 sB[64 * 32];  // [n][k]
  int tid = threadIdx.x, lane = tid & 31, w = tid >> 5;
  int mtiles = M / 64;
  int m0 = (blockIdx.x % mtiles) * 64;
  int n0 = (blockIdx.x / mtiles) * 64;
  int mt = w & 3;
  int ntb = (w >> 2) * 2;
  int r = tid >> 2, part = tid & 3;  // staging coords: one b128 each
  const __bf16* ga = A + (size_t)(m0 + r) * lda + part * 8;
  const __bf16* gb = Bt + (size_t)(n0 + r) * ldk + part * 8;
  unsigned ldsA = (unsigned)(uintptr_t)(sA + r * 32 + part * 8);
  unsigned ldsB = (unsigned)(uintptr_t)(sB + r * 32 + part * 8);
  f32x8 acc[2] = {fzero8(), fzero8()};

  for (int k0 = 0; k0 < Kp; k0 += 32) {
    __syncthreads();
    async_copy_b128(ldsA, ga + k0);
    async_copy_b128(ldsB, gb + k0);
    if (k0 + 128 < Kp) {  // gfx1250 global_prefetch_b8 ahead of the K walk
      __builtin_prefetch(ga + k0 + 128, 0, 1);
      __builtin_prefetch(gb + k0 + 128, 0, 1);
    }
    wait_asynccnt0();
    __syncthreads();
    bf16x16 a = load_a_frag(sA, 32, mt * 16, 0, lane);
#pragma unroll
    for (int j = 0; j < 2; ++j) {
      bf16x16 b = load_bt_frag(sB, 32, 0, (ntb + j) * 16, lane);
      acc[j] = wmma_bf16(a, b, acc[j]);
    }
  }
#pragma unroll
  for (int j = 0; j < 2; ++j) {
    int col = n0 + (ntb + j) * 16 + (lane & 15);
#pragma unroll
    for (int r8 = 0; r8 < 8; ++r8) {
      int row = m0 + mt * 16 + r8 + ((lane >> 4) << 3);
      C[(size_t)row * N + col] = acc[j][r8];
    }
  }
}

// BatchNorm(training stats) + ReLU, fp32 in -> bf16 out. One block / column.
// (FC bias shifts cancel inside BN, so they are omitted upstream.)
__global__ __launch_bounds__(256) void bn_relu(const float* __restrict__ H,
                                               const float* __restrict__ gam,
                                               const float* __restrict__ bet,
                                               __bf16* __restrict__ out,
                                               int rows, int cols) {
  int c = blockIdx.x, tid = threadIdx.x;
  __shared__ float ssum[256], ssq[256];
  float s = 0.0f, q = 0.0f;
  for (int r = tid; r < rows; r += 256) {
    float v = H[(size_t)r * cols + c];
    s += v;
    q += v * v;
  }
  ssum[tid] = s;
  ssq[tid] = q;
  __syncthreads();
  for (int st = 128; st > 0; st >>= 1) {
    if (tid < st) {
      ssum[tid] += ssum[tid + st];
      ssq[tid] += ssq[tid + st];
    }
    __syncthreads();
  }
  float mean = ssum[0] / rows;
  float var = ssq[0] / rows - mean * mean;
  float scale = rsqrtf(var + BNEPS) * gam[c];
  float shift = bet[c] - mean * scale;
  for (int r = tid; r < rows; r += 256) {
    float v = H[(size_t)r * cols + c] * scale + shift;
    out[(size_t)r * cols + c] = (__bf16)fmaxf(v, 0.0f);
  }
}

// FC3 (256 -> 3) + bias + softmax, one thread per batch row.
__global__ void fc3_softmax(const __bf16* __restrict__ h,
                            const float* __restrict__ W,
                            const float* __restrict__ b,
                            float* __restrict__ out, int rows) {
  int r = blockIdx.x * blockDim.x + threadIdx.x;
  if (r >= rows) return;
  float a0 = b[0], a1 = b[1], a2 = b[2];
  const __bf16* hp = h + (size_t)r * LIN2;
  for (int k = 0; k < LIN2; ++k) {
    float v = (float)hp[k];
    a0 += v * W[k * 3 + 0];
    a1 += v * W[k * 3 + 1];
    a2 += v * W[k * 3 + 2];
  }
  float m = fmaxf(a0, fmaxf(a1, a2));
  float e0 = __expf(a0 - m), e1 = __expf(a1 - m), e2 = __expf(a2 - m);
  float si = 1.0f / (e0 + e1 + e2);
  out[(size_t)r * 3 + 0] = e0 * si;
  out[(size_t)r * 3 + 1] = e1 * si;
  out[(size_t)r * 3 + 2] = e2 * si;
}

// ---------------------------------------------------------------------------
extern "C" void kernel_launch(void* const* d_in, const int* in_sizes, int n_in,
                              void* d_out, int out_size, void* d_ws,
                              size_t ws_size, hipStream_t stream) {
  const float* x   = (const float*)d_in[0];
  const int*   ei  = (const int*)d_in[1];
  const float* W1  = (const float*)d_in[2];
  const float* b1  = (const float*)d_in[3];
  const float* W2  = (const float*)d_in[4];
  const float* b2  = (const float*)d_in[5];
  const float* W3  = (const float*)d_in[6];
  const float* b3  = (const float*)d_in[7];
  const float* W4  = (const float*)d_in[8];
  const float* b4  = (const float*)d_in[9];
  const float* Wh1 = (const float*)d_in[10];
  const float* g1  = (const float*)d_in[12];
  const float* be1 = (const float*)d_in[13];
  const float* Wh2 = (const float*)d_in[14];
  const float* g2  = (const float*)d_in[16];
  const float* be2 = (const float*)d_in[17];
  const float* Wh3 = (const float*)d_in[18];
  const float* bh3 = (const float*)d_in[19];
  (void)n_in; (void)out_size; (void)ws_size;

  int E = in_sizes[1] / 2;

  char* p = (char*)d_ws;
  auto alloc = [&](size_t bytes) -> char* {
    char* r = p;
    p += (bytes + 255) & ~(size_t)255;
    return r;
  };
  float*  deg   = (float*)alloc((size_t)NTOT * 4);
  float*  lhat  = (float*)alloc(64 * 64 * 4);
  __bf16* xcat  = (__bf16*)alloc((size_t)NTOT * FEAT * 2);
  __bf16* featp = (__bf16*)alloc((size_t)BATCHG * FLATKP * 2);
  __bf16* wt1   = (__bf16*)alloc((size_t)3 * NODESP * 128 * 2);
  __bf16* wt2   = (__bf16*)alloc((size_t)3 * NODESP * 192 * 2);
  __bf16* wt3   = (__bf16*)alloc((size_t)3 * NODESP * 256 * 2);
  __bf16* wt4   = (__bf16*)alloc((size_t)3 * NODESP * 320 * 2);
  __bf16* wh1t  = (__bf16*)alloc((size_t)LIN1 * FLATKP * 2);
  __bf16* wh2t  = (__bf16*)alloc((size_t)LIN2 * LIN1 * 2);
  float*  h1    = (float*)alloc((size_t)BATCHG * LIN1 * 4);
  __bf16* h1b   = (__bf16*)alloc((size_t)BATCHG * LIN1 * 2);
  float*  h2    = (float*)alloc((size_t)BATCHG * LIN2 * 4);
  __bf16* h2b   = (__bf16*)alloc((size_t)BATCHG * LIN2 * 2);

  // graph structure -> shared 62x62 scaled Laplacian
  zero_f32<<<512, 256, 0, stream>>>(deg, NTOT);
  zero_f32<<<16, 256, 0, stream>>>(lhat, 64 * 64);
  deg_accum<<<(E + 255) / 256, 256, 0, stream>>>(ei, deg, E);
  lhat_build<<<(E + 255) / 256, 256, 0, stream>>>(ei, deg, lhat, E);

  // weight transpose + bf16 conversion (N and K zero-padded)
  wprep<<<256, 256, 0, stream>>>(W1, wt1, 3, 128, NODES, NODESP, 128);
  wprep<<<256, 256, 0, stream>>>(W2, wt2, 3, 190, NODES, NODESP, 192);
  wprep<<<256, 256, 0, stream>>>(W3, wt3, 3, 252, NODES, NODESP, 256);
  wprep<<<256, 256, 0, stream>>>(W4, wt4, 3, 314, NODES, NODESP, 320);
  wprep<<<2048, 256, 0, stream>>>(Wh1, wh1t, 1, FLATK, LIN1, LIN1, FLATKP);
  wprep<<<256, 256, 0, stream>>>(Wh2, wh2t, 1, LIN1, LIN2, LIN2, LIN1);

  // 4 Cheb layers with dense skip concat (one workgroup per graph)
  cheb_kernel<<<BATCHG, 256, 0, stream>>>(x, xcat, lhat, wt1, b1, 128, 0);
  cheb_kernel<<<BATCHG, 256, 0, stream>>>(x, xcat, lhat, wt2, b2, 192, 62);
  cheb_kernel<<<BATCHG, 256, 0, stream>>>(x, xcat, lhat, wt3, b3, 256, 124);
  cheb_kernel<<<BATCHG, 256, 0, stream>>>(x, xcat, lhat, wt4, b4, 320, 186);

  // FC1: pad feats rows to K=15392, then guard-free WMMA GEMM -> BN -> ReLU
  pad_feats<<<2048, 256, 0, stream>>>(xcat, featp);
  gemm_bf16<<<(BATCHG / 64) * (LIN1 / 64), 256, 0, stream>>>(
      featp, FLATKP, wh1t, FLATKP, h1, BATCHG, LIN1, FLATKP);
  bn_relu<<<LIN1, 256, 0, stream>>>(h1, g1, be1, h1b, BATCHG, LIN1);

  // FC2 -> BN -> ReLU
  gemm_bf16<<<(BATCHG / 64) * (LIN2 / 64), 256, 0, stream>>>(
      h1b, LIN1, wh2t, LIN1, h2, BATCHG, LIN2, LIN1);
  bn_relu<<<LIN2, 256, 0, stream>>>(h2, g2, be2, h2b, BATCHG, LIN2);

  // FC3 + softmax
  fc3_softmax<<<BATCHG / 256, 256, 0, stream>>>(h2b, Wh3, bh3, (float*)d_out,
                                                BATCHG);
}
